// RecursiveScaledDotProductAttention_29128468201799
// MI455X (gfx1250) — matrix operand order repaired and checked by reading, not compiled, not run on hardware
//
#include <hip/hip_runtime.h>

// Problem constants (match reference)
#define BB 4
#define SS 2048
#define DD 1024
#define HH 16
#define HDI 64
#define MM (BB * SS)

typedef __attribute__((ext_vector_type(16))) __bf16 v16bf;
typedef __attribute__((ext_vector_type(8)))  __bf16 bf16x8;
typedef __attribute__((ext_vector_type(8)))  float  v8f;
typedef __attribute__((ext_vector_type(4)))  float  f32x4;

__device__ __forceinline__ v8f wmma_bf16(v16bf a, v16bf b, v8f c) {
  // D = A(16x32 bf16) * B(32x16 bf16) + C(16x16 f32)
  return __builtin_amdgcn_wmma_f32_16x16x32_bf16(false, a, false, b, (short)0, c,
                                                 false, false);
}

__device__ __forceinline__ v16bf join8(bf16x8 lo, bf16x8 hi) {
  v16bf r;
#pragma unroll
  for (int j = 0; j < 8; ++j) { r[j] = lo[j]; r[8 + j] = hi[j]; }
  return r;
}

__device__ __forceinline__ v16bf cvt16(f32x4 a, f32x4 b, f32x4 c, f32x4 d) {
  v16bf r;
#pragma unroll
  for (int j = 0; j < 4; ++j) {
    r[j]      = (__bf16)a[j];
    r[4 + j]  = (__bf16)b[j];
    r[8 + j]  = (__bf16)c[j];
    r[12 + j] = (__bf16)d[j];
  }
  return r;
}

// ---------------------------------------------------------------------------
// One-time weight prep: Wt[n][k] = bf16(W[k][n]).  Tiled LDS transpose.
// Grid: (N/32, K/32), block 256.
// ---------------------------------------------------------------------------
__global__ __launch_bounds__(256) void transpose_w(const float* __restrict__ in,
                                                   __bf16* __restrict__ out,
                                                   int K, int N) {
  __shared__ float tile[32][33];
  const int k0 = blockIdx.y * 32, n0 = blockIdx.x * 32;
  const int tn = threadIdx.x & 31, tk = threadIdx.x >> 5;  // 8 k-rows per pass
#pragma unroll
  for (int r = 0; r < 32; r += 8)
    tile[tk + r][tn] = in[(size_t)(k0 + tk + r) * N + n0 + tn];
  __syncthreads();
#pragma unroll
  for (int r = 0; r < 32; r += 8)
    out[(size_t)(n0 + tk + r) * K + k0 + tn] = (__bf16)tile[tn][tk + r];
}

// ---------------------------------------------------------------------------
// Tiled WMMA GEMM, weights pre-transposed bf16 [N][Ktot] (B-frags are two
// contiguous global b128 loads from L2).  Block = 256 threads (8 waves),
// tile 128(M) x 128(N), no LDS, no barriers.
// MODE 0: Y(bf16) = A(f32) @ W                      (Q/K projections)
// MODE 3: like 0 but output scattered transposed:   V -> vT[B,H,HD,S]
// MODE 1: Y(f32)  = A(bf16) @ W + bias              (output projection)
// MODE 2: gate: g = sigmoid([cur|out] @ Wg + bg); dst = cur*(1-g) + out*g
// ---------------------------------------------------------------------------
template <int MODE>
__global__ __launch_bounds__(256) void gemm_wmma_kernel(
    const float* __restrict__ Af, const __bf16* __restrict__ Abf,
    const float* __restrict__ A2f, const __bf16* __restrict__ Wt,
    const float* __restrict__ bias, __bf16* __restrict__ Ybf,
    float* __restrict__ Yf, const float* __restrict__ curIn,
    const float* __restrict__ outIn, float* __restrict__ dstF, int M, int N,
    int Ktot) {
  const int tid  = threadIdx.x;
  const int wave = tid >> 5;
  const int lane = tid & 31;
  const int nl   = lane & 15;
  const int hh   = lane >> 4;
  const int m0   = blockIdx.y * 128 + wave * 16;
  const int n0   = blockIdx.x * 128;

  v8f cacc[8];
#pragma unroll
  for (int nt = 0; nt < 8; ++nt)
#pragma unroll
    for (int r = 0; r < 8; ++r) cacc[nt][r] = 0.0f;

  for (int k0 = 0; k0 < Ktot; k0 += 32) {
    // A fragment, 16x32: lane m = nl, element i -> k = 8*hh + (i&7) + 16*(i>>3)
    v16bf a;
    if (MODE == 1) {
      const __bf16* ap = Abf + (size_t)(m0 + nl) * DD + k0 + hh * 8;
      a = join8(*(const bf16x8*)ap, *(const bf16x8*)(ap + 16));
      __builtin_prefetch((const void*)(ap + 32), 0, 0);
    } else {
      const float* abase = Af;
      int kk = k0;
      if (MODE == 2 && k0 >= DD) { abase = A2f; kk = k0 - DD; }
      const float* ap = abase + (size_t)(m0 + nl) * DD + kk + hh * 8;
      a = cvt16(*(const f32x4*)ap, *(const f32x4*)(ap + 4),
                *(const f32x4*)(ap + 16), *(const f32x4*)(ap + 20));
      __builtin_prefetch((const void*)(ap + 32), 0, 0);
    }

    // B fragments: lane n = nl, element i -> k = 8*hh + (i&7) + 16*(i>>3)
#pragma unroll
    for (int nt = 0; nt < 8; ++nt) {
      const __bf16* bp = Wt + (size_t)(n0 + nt * 16 + nl) * Ktot + k0 + hh * 8;
      const v16bf b = join8(*(const bf16x8*)bp, *(const bf16x8*)(bp + 16));
      cacc[nt] = wmma_bf16(a, b, cacc[nt]);
    }
  }

  // Epilogue. D layout: lane col = nl + 16*nt, VGPR r -> row r + 8*hh.
#pragma unroll
  for (int nt = 0; nt < 8; ++nt) {
    const int col = n0 + nt * 16 + nl;
#pragma unroll
    for (int r = 0; r < 8; ++r) {
      const int m = m0 + r + 8 * hh;
      if (MODE == 0) {
        Ybf[(size_t)m * N + col] = (__bf16)cacc[nt][r];
      } else if (MODE == 3) {
        // vT[b][h][d][s] = value at (token m, feature col)
        const int b = m >> 11, s = m & (SS - 1);
        const int h = col >> 6, d = col & (HDI - 1);
        Ybf[((size_t)(b * HH + h) * HDI + d) * SS + s] = (__bf16)cacc[nt][r];
      } else if (MODE == 1) {
        Yf[(size_t)m * N + col] = cacc[nt][r] + bias[col];
      } else {
        const size_t idx = (size_t)m * N + col;
        const float pre = cacc[nt][r] + bias[col];
        const float g   = 1.0f / (1.0f + __expf(-pre));
        const float cv  = curIn[idx];
        const float ov  = outIn[idx];
        dstF[idx] = cv + g * (ov - cv);
      }
    }
  }
}

// ---------------------------------------------------------------------------
// Attention: one wave per (b, h, 16-query tile). Full 16x2048 score block in
// LDS (bf16, 64 KB), in-LDS softmax, then ctx = P @ V with WMMA.
// q/k layout: [B, S, H, HD] bf16;  v layout: TRANSPOSED [B, H, HD, S] bf16.
// ---------------------------------------------------------------------------
__global__ __launch_bounds__(32) void attn_kernel(
    const __bf16* __restrict__ q, const __bf16* __restrict__ k,
    const __bf16* __restrict__ vT, const float* __restrict__ obs,
    __bf16* __restrict__ ctx, float* __restrict__ attnOut, int writeAttn) {
  __shared__ __bf16 Ssc[16 * SS];  // 64 KB
  const int lane = threadIdx.x;
  const int nl   = lane & 15;
  const int hh   = lane >> 4;
  const int q0   = blockIdx.x * 16;
  const int h    = blockIdx.y;
  const int b    = blockIdx.z;
  const float obsv = obs[b * HH + h];

  // Q fragments (A-matrix): rows q0..q0+15, d-halves [0,32) and [32,64)
  const __bf16* qp = q + (size_t)(b * SS + q0 + nl) * DD + h * HDI;
  const v16bf a0 =
      join8(*(const bf16x8*)(qp + hh * 8), *(const bf16x8*)(qp + 16 + hh * 8));
  const v16bf a1 =
      join8(*(const bf16x8*)(qp + 32 + hh * 8), *(const bf16x8*)(qp + 48 + hh * 8));

  // scores[qi][kj] = (q . k) * scale + obs -> LDS
  for (int kt = 0; kt < SS / 16; ++kt) {
    const __bf16* kp = k + (size_t)(b * SS + kt * 16 + nl) * DD + h * HDI;
    const v16bf b0 =
        join8(*(const bf16x8*)(kp + hh * 8), *(const bf16x8*)(kp + 16 + hh * 8));
    const v16bf b1 = join8(*(const bf16x8*)(kp + 32 + hh * 8),
                           *(const bf16x8*)(kp + 48 + hh * 8));
    v8f acc;
#pragma unroll
    for (int r = 0; r < 8; ++r) acc[r] = 0.0f;
    acc = wmma_bf16(a0, b0, acc);
    acc = wmma_bf16(a1, b1, acc);
#pragma unroll
    for (int r = 0; r < 8; ++r)
      Ssc[(r + 8 * hh) * SS + kt * 16 + nl] = (__bf16)(acc[r] * 0.125f + obsv);
  }
  __syncthreads();

  // Row-wise softmax over LDS (wave-cooperative, shfl_xor reductions)
  for (int m = 0; m < 16; ++m) {
    __bf16* row = &Ssc[m * SS];
    float mx = -3.0e38f;
    for (int j = lane; j < SS; j += 32) mx = fmaxf(mx, (float)row[j]);
#pragma unroll
    for (int o = 16; o > 0; o >>= 1) mx = fmaxf(mx, __shfl_xor(mx, o, 32));
    float sum = 0.0f;
    for (int j = lane; j < SS; j += 32) {
      const float e = __expf((float)row[j] - mx);
      row[j] = (__bf16)e;
      sum += e;
    }
#pragma unroll
    for (int o = 16; o > 0; o >>= 1) sum += __shfl_xor(sum, o, 32);
    const float inv = 1.0f / sum;
    for (int j = lane; j < SS; j += 32) row[j] = (__bf16)((float)row[j] * inv);
  }
  __syncthreads();

  // Final iteration: stream attention probabilities out as f32
  if (writeAttn) {
    for (int m = 0; m < 16; ++m) {
      float* drow = attnOut + ((size_t)(b * HH + h) * SS + q0 + m) * SS;
      const bf16x8* srow = (const bf16x8*)&Ssc[m * SS];
      for (int j = lane; j < SS / 8; j += 32) {
        const bf16x8 e = srow[j];
        f32x4 lo, hi;
#pragma unroll
        for (int t = 0; t < 4; ++t) { lo[t] = (float)e[t]; hi[t] = (float)e[4 + t]; }
        *(f32x4*)(drow + j * 8)     = lo;
        *(f32x4*)(drow + j * 8 + 4) = hi;
      }
    }
  }

  // ctx = P @ V : A = probs (16 x 2048) from LDS, B frag = 2 contiguous b128
  // loads from transposed vT[b][h][d][kj].
  v8f cacc[4];
#pragma unroll
  for (int nt = 0; nt < 4; ++nt)
#pragma unroll
    for (int r = 0; r < 8; ++r) cacc[nt][r] = 0.0f;

  const __bf16* vhead = vT + (size_t)(b * HH + h) * HDI * SS;
  for (int ks = 0; ks < SS; ks += 32) {
    const __bf16* pr = &Ssc[nl * SS + ks + hh * 8];
    const v16bf a = join8(*(const bf16x8*)pr, *(const bf16x8*)(pr + 16));
#pragma unroll
    for (int nt = 0; nt < 4; ++nt) {
      const int d = nt * 16 + nl;
      const __bf16* vp = vhead + (size_t)d * SS + ks + hh * 8;
      const v16bf bb = join8(*(const bf16x8*)vp, *(const bf16x8*)(vp + 16));
      cacc[nt] = wmma_bf16(a, bb, cacc[nt]);
    }
  }

#pragma unroll
  for (int nt = 0; nt < 4; ++nt)
#pragma unroll
    for (int r = 0; r < 8; ++r)
      ctx[(size_t)(b * SS + q0 + r + 8 * hh) * DD + h * HDI + nt * 16 + nl] =
          (__bf16)cacc[nt][r];
}

// obs[b,h] = observer_state[b,:] . Wobs[:,h] + bobs[h]
__global__ __launch_bounds__(256) void obs_kernel(const float* __restrict__ os,
                                                  const float* __restrict__ Wobs,
                                                  const float* __restrict__ bobs,
                                                  float* __restrict__ obsOut) {
  const int h = blockIdx.x, b = blockIdx.y;
  __shared__ float red[256];
  float s = 0.0f;
  for (int d = threadIdx.x; d < DD; d += 256) s += os[b * DD + d] * Wobs[d * HH + h];
  red[threadIdx.x] = s;
  __syncthreads();
  for (int o = 128; o > 0; o >>= 1) {
    if (threadIdx.x < o) red[threadIdx.x] += red[threadIdx.x + o];
    __syncthreads();
  }
  if (threadIdx.x == 0) obsOut[b * HH + h] = red[0] + bobs[h];
}

__global__ __launch_bounds__(256) void copy_f32(const float* __restrict__ src,
                                                float* __restrict__ dst,
                                                size_t n4) {
  const size_t i = (size_t)blockIdx.x * 256 + threadIdx.x;
  if (i < n4) ((f32x4*)dst)[i] = ((const f32x4*)src)[i];
}

// ---------------------------------------------------------------------------
extern "C" void kernel_launch(void* const* d_in, const int* in_sizes, int n_in,
                              void* d_out, int out_size, void* d_ws,
                              size_t ws_size, hipStream_t stream) {
  const float* x    = (const float*)d_in[0];
  const float* obsS = (const float*)d_in[1];
  const float* Wq   = (const float*)d_in[2];
  const float* Wk   = (const float*)d_in[3];
  const float* Wv   = (const float*)d_in[4];
  const float* Wo   = (const float*)d_in[5];
  const float* bo   = (const float*)d_in[6];
  const float* Wg   = (const float*)d_in[7];
  const float* bg   = (const float*)d_in[8];
  const float* Wobs = (const float*)d_in[9];
  const float* bobs = (const float*)d_in[10];

  // Workspace layout (bytes):
  // cur(f32) | q | k | vT | ctx (bf16) | out(f32) | obs | Wt's (bf16)
  char* ws = (char*)d_ws;
  float*  curW = (float*)(ws + 0);                  // 33,554,432
  __bf16* qb   = (__bf16*)(ws + 33554432);          // 16,777,216
  __bf16* kb   = (__bf16*)(ws + 50331648);          // 16,777,216
  __bf16* vTb  = (__bf16*)(ws + 67108864);          // 16,777,216 ([B,H,HD,S])
  __bf16* cb   = (__bf16*)(ws + 83886080);          // 16,777,216
  float*  outb = (float*)(ws + 100663296);          // 33,554,432
  float*  obsb = (float*)(ws + 134217728);          // 4,096 (padded)
  __bf16* Wqt  = (__bf16*)(ws + 134221824);         // 2,097,152  [N=1024][K=1024]
  __bf16* Wkt  = (__bf16*)(ws + 136318976);         // 2,097,152
  __bf16* Wvt  = (__bf16*)(ws + 138416128);         // 2,097,152
  __bf16* Wot  = (__bf16*)(ws + 140513280);         // 2,097,152
  __bf16* Wgt  = (__bf16*)(ws + 142610432);         // 4,194,304  [N=1024][K=2048]

  float* curOut  = (float*)d_out;             // [B*S*D]
  float* attnOut = curOut + (size_t)MM * DD;  // [B*H*S*S]

  // One-time prep: weight transposes + bf16 convert; x copy; observer bias.
  transpose_w<<<dim3(32, 32), 256, 0, stream>>>(Wq, Wqt, DD, DD);
  transpose_w<<<dim3(32, 32), 256, 0, stream>>>(Wk, Wkt, DD, DD);
  transpose_w<<<dim3(32, 32), 256, 0, stream>>>(Wv, Wvt, DD, DD);
  transpose_w<<<dim3(32, 32), 256, 0, stream>>>(Wo, Wot, DD, DD);
  transpose_w<<<dim3(32, 64), 256, 0, stream>>>(Wg, Wgt, 2 * DD, DD);
  copy_f32<<<(MM * DD / 4 + 255) / 256, 256, 0, stream>>>(x, curW,
                                                          (size_t)MM * DD / 4);
  obs_kernel<<<dim3(HH, BB), 256, 0, stream>>>(obsS, Wobs, bobs, obsb);

  float* src = curW;
  float* dst = curOut;
  const dim3 gg(DD / 128, MM / 128);  // (8, 64)
  for (int it = 0; it < 3; ++it) {
    gemm_wmma_kernel<0><<<gg, 256, 0, stream>>>(src, nullptr, nullptr, Wqt,
                                                nullptr, qb, nullptr, nullptr,
                                                nullptr, nullptr, MM, DD, DD);
    gemm_wmma_kernel<0><<<gg, 256, 0, stream>>>(src, nullptr, nullptr, Wkt,
                                                nullptr, kb, nullptr, nullptr,
                                                nullptr, nullptr, MM, DD, DD);
    gemm_wmma_kernel<3><<<gg, 256, 0, stream>>>(src, nullptr, nullptr, Wvt,
                                                nullptr, vTb, nullptr, nullptr,
                                                nullptr, nullptr, MM, DD, DD);
    attn_kernel<<<dim3(SS / 16, HH, BB), 32, 0, stream>>>(qb, kb, vTb, obsb, cb,
                                                          attnOut,
                                                          (it == 2) ? 1 : 0);
    gemm_wmma_kernel<1><<<gg, 256, 0, stream>>>(nullptr, cb, nullptr, Wot, bo,
                                                nullptr, outb, nullptr, nullptr,
                                                nullptr, MM, DD, DD);
    gemm_wmma_kernel<2><<<gg, 256, 0, stream>>>(src, nullptr, outb, Wgt, bg,
                                                nullptr, nullptr, src, outb,
                                                dst, MM, DD, 2 * DD);
    float* t = src;
    src = dst;
    dst = t;
  }
  // after 3 iterations the final `cur` lands in d_out (curOut)
}
